// MultiHeadAttentionLayer_52845277610306
// MI455X (gfx1250) — compile-verified, matrix operand order checked by reading, not run on hardware
//
#include <hip/hip_runtime.h>
#include <hip/hip_bf16.h>

typedef __attribute__((ext_vector_type(16))) _Float16 v16h;
typedef __attribute__((ext_vector_type(8)))  _Float16 v8h;
typedef __attribute__((ext_vector_type(4)))  _Float16 v4h;
typedef __attribute__((ext_vector_type(8)))  float    v8f;
typedef __attribute__((ext_vector_type(4)))  float    v4f;
typedef __attribute__((ext_vector_type(4)))  int      v4i;

#define NHEADS 16
#define HEAD   64
#define DM     1024
#define LSEQ   2048
#define BATCH  2

// ---------------------------------------------------------------------------
// Fragment helpers (ISA 7.12.2, 16-bit 16x32 A-matrix):
// lane L holds M = L&15; elements 0..7 = K = 8*(L>>4)+0..7,
// elements 8..15 = K = 16+8*(L>>4)+0..7.  B mirrors A with lane = N, so for
// X @ W^T both operands load contiguous rows.  `p` -> lane's first 8 elems.
// ---------------------------------------------------------------------------
static __device__ __forceinline__ v16h frag_f16(const _Float16* p) {
  v8h lo = *(const v8h*)(p);
  v8h hi = *(const v8h*)(p + 16);
  v16h f;
#pragma unroll
  for (int i = 0; i < 8; ++i) { f[i] = lo[i]; f[8 + i] = hi[i]; }
  return f;
}

static __device__ __forceinline__ v8f wmma_f16(v16h a, v16h b, v8f c) {
  return __builtin_amdgcn_wmma_f32_16x16x32_f16(false, a, false, b,
                                                (short)0, c, false, false);
}

static __device__ __forceinline__ v8f vzero8() {
  v8f z = {0.f, 0.f, 0.f, 0.f, 0.f, 0.f, 0.f, 0.f};
  return z;
}

// ---------------------------------------------------------------------------
// One-shot fp32 -> f16 cast (8 elements/thread, vectorized).
// ---------------------------------------------------------------------------
__global__ void cast_f16(const float* __restrict__ src, _Float16* __restrict__ dst,
                         int n8) {
  int i = blockIdx.x * blockDim.x + threadIdx.x;
  if (i >= n8) return;
  const float* p = src + (size_t)i * 8;
  v4f a0 = *(const v4f*)(p);
  v4f a1 = *(const v4f*)(p + 4);
  v8h o;
#pragma unroll
  for (int k = 0; k < 4; ++k) { o[k] = (_Float16)a0[k]; o[4 + k] = (_Float16)a1[k]; }
  *(v8h*)(dst + (size_t)i * 8) = o;
}

// ---------------------------------------------------------------------------
// C = X @ W^T + bias, all-f16 operands, fp32 accumulate.
// M=4096, N=1024, K=1024.  Block = 128 thr (4 waves); wave owns a 32x64 tile
// (8 independent accumulators -> deep WMMA pipelining); grid (32,16).
// EPI 0: per-head L2-normalize rows, f16 [B,H,L,64]   (Q, K)
// EPI 1: f16 transposed [B,H,64,L]                    (V)
// EPI 2: fp32 [M,DM] + bias                           (output projection)
// ---------------------------------------------------------------------------
template <int EPI>
__global__ void gemm_xwT(const _Float16* __restrict__ X, const _Float16* __restrict__ W,
                         const float* __restrict__ bias,
                         _Float16* __restrict__ out16, float* __restrict__ out32) {
  const int lane = threadIdx.x & 31;
  const int wv   = threadIdx.x >> 5;
  const int r    = lane & 15;
  const int hi   = lane >> 4;
  const int row0 = blockIdx.x * 128 + wv * 32;
  const int n0   = blockIdx.y * 64;

  v8f acc[2][4];
#pragma unroll
  for (int mt = 0; mt < 2; ++mt)
#pragma unroll
    for (int t = 0; t < 4; ++t) acc[mt][t] = vzero8();

  const _Float16* x0 = X + (size_t)(row0 + r) * DM + hi * 8;
  const _Float16* x1 = x0 + (size_t)16 * DM;
  for (int k0 = 0; k0 < DM; k0 += 32) {
    if (k0 + 32 < DM) {                       // gfx1250 global_prefetch_b8
      __builtin_prefetch(x0 + k0 + 32, 0, 3);
      __builtin_prefetch(W + (size_t)(n0 + r) * DM + k0 + 32, 0, 3);
    }
    v16h a0 = frag_f16(x0 + k0);
    v16h a1 = frag_f16(x1 + k0);
#pragma unroll
    for (int t = 0; t < 4; ++t) {
      v16h b = frag_f16(W + (size_t)(n0 + t * 16 + r) * DM + k0 + hi * 8);
      acc[0][t] = wmma_f16(a0, b, acc[0][t]);
      acc[1][t] = wmma_f16(a1, b, acc[1][t]);
    }
  }

#pragma unroll
  for (int t = 0; t < 4; ++t) {
    float bv = bias[n0 + t * 16 + r];
#pragma unroll
    for (int mt = 0; mt < 2; ++mt)
#pragma unroll
      for (int v = 0; v < 8; ++v) acc[mt][t][v] += bv;
  }

  if (EPI == 0) {            // L2-normalize over head dim (N-tile == one head)
    const int h = blockIdx.y;
#pragma unroll
    for (int mt = 0; mt < 2; ++mt)
#pragma unroll
      for (int v = 0; v < 8; ++v) {
        float s = 0.f;
#pragma unroll
        for (int t = 0; t < 4; ++t) s += acc[mt][t][v] * acc[mt][t][v];
        s += __shfl_xor(s, 1);
        s += __shfl_xor(s, 2);
        s += __shfl_xor(s, 4);
        s += __shfl_xor(s, 8);           // lanes sharing row M = v + 8*hi
        float inv = 1.0f / fmaxf(sqrtf(s), 1e-12f);
        int m  = row0 + mt * 16 + v + 8 * hi;
        int bb = m >> 11;                // / LSEQ
        int l  = m & (LSEQ - 1);
        _Float16* dst = out16 + (((size_t)bb * NHEADS + h) * LSEQ + l) * HEAD;
#pragma unroll
        for (int t = 0; t < 4; ++t) dst[t * 16 + r] = (_Float16)(acc[mt][t][v] * inv);
      }
  } else if (EPI == 1) {     // V transposed per head: [B,H,64,L]
    const int h = blockIdx.y;
#pragma unroll
    for (int mt = 0; mt < 2; ++mt)
#pragma unroll
      for (int v = 0; v < 8; ++v) {
        int m  = row0 + mt * 16 + v + 8 * hi;
        int bb = m >> 11;
        int l  = m & (LSEQ - 1);
#pragma unroll
        for (int t = 0; t < 4; ++t)
          out16[(((size_t)bb * NHEADS + h) * HEAD + t * 16 + r) * LSEQ + l] =
              (_Float16)acc[mt][t][v];
      }
  } else {                   // fp32 final output
#pragma unroll
    for (int mt = 0; mt < 2; ++mt)
#pragma unroll
      for (int v = 0; v < 8; ++v) {
        int m = row0 + mt * 16 + v + 8 * hi;
#pragma unroll
        for (int t = 0; t < 4; ++t)
          out32[(size_t)m * DM + n0 + t * 16 + r] = acc[mt][t][v];
      }
  }
}

// ---------------------------------------------------------------------------
// Fused attention: block = (b, h, 16 query rows), 128 threads (4 waves).
// S = scale * Qn Kn^T -> LDS f32 (16 x 2048, pitch 2056); masked softmax in
// LDS (float4-vectorized: 8 thr/row x 4 elems -> 128B contiguous per row per
// step, attn written once with global_store_b128); probs also written to a
// second LDS region as f16 (pitch 2064) so the P@V loop is conversion-free.
// ~193 KB LDS/WG — only possible on CDNA5's 320 KB WGP LDS.
// ---------------------------------------------------------------------------
#define LDP  (LSEQ + 8)     // f32 pitch
#define LDPH (LSEQ + 16)    // f16 pitch
#define SMEM_BYTES ((size_t)16 * LDP * 4 + (size_t)16 * LDPH * 2)

__global__ void attn_fused(const _Float16* __restrict__ Qh, const _Float16* __restrict__ Kh,
                           const _Float16* __restrict__ Vt, const int* __restrict__ mask,
                           const float* __restrict__ scale_p,
                           float* __restrict__ attn, _Float16* __restrict__ ctx16) {
  extern __shared__ char smem[];
  float*    Sl = (float*)smem;                            // 16 x 2056 f32
  _Float16* Pl = (_Float16*)(smem + (size_t)16 * LDP * 4);// 16 x 2064 f16
  const int lane = threadIdx.x & 31;
  const int wv   = threadIdx.x >> 5;
  const int r    = lane & 15;
  const int hi   = lane >> 4;
  const int b  = blockIdx.z;
  const int h  = blockIdx.y;
  const int q0 = blockIdx.x * 16;
  const size_t bh = (size_t)b * NHEADS + h;
  const _Float16* Qb = Qh + bh * (size_t)LSEQ * HEAD;
  const _Float16* Kb = Kh + bh * (size_t)LSEQ * HEAD;
  const _Float16* Vb = Vt + bh * (size_t)HEAD * LSEQ;
  const float scale = scale_p[0];

  // ---- scores: S[16 x 2048] ----
  v16h a0 = frag_f16(Qb + (size_t)(q0 + r) * HEAD + hi * 8);
  v16h a1 = frag_f16(Qb + (size_t)(q0 + r) * HEAD + 32 + hi * 8);
  for (int nt = wv; nt < LSEQ / 16; nt += 4) {
    const _Float16* kb = Kb + (size_t)(nt * 16 + r) * HEAD + hi * 8;
    v8f acc = vzero8();
    acc = wmma_f16(a0, frag_f16(kb), acc);
    acc = wmma_f16(a1, frag_f16(kb + 32), acc);
#pragma unroll
    for (int v = 0; v < 8; ++v)
      Sl[(v + 8 * hi) * LDP + nt * 16 + r] = scale * acc[v];
  }
  __syncthreads();

  // ---- masked softmax: 8 threads/row, float4 per thread ----
  {
    const int row = threadIdx.x >> 3;
    const int sub = threadIdx.x & 7;
    const int* mrow = mask + (size_t)b * LSEQ;    // mask shape [B,1,1,L]
    float*    srow = Sl + row * LDP;
    _Float16* prow = Pl + row * LDPH;

    float mx = -3.0e38f;
    for (int j = sub * 4; j < LSEQ; j += 32) {
      v4f v = *(v4f*)(srow + j);
      v4i m = *(const v4i*)(mrow + j);
#pragma unroll
      for (int c = 0; c < 4; ++c) {
        if (m[c] == 0) v[c] = -1.0e18f;
        mx = fmaxf(mx, v[c]);
      }
      *(v4f*)(srow + j) = v;
    }
    mx = fmaxf(mx, __shfl_xor(mx, 1));
    mx = fmaxf(mx, __shfl_xor(mx, 2));
    mx = fmaxf(mx, __shfl_xor(mx, 4));

    float sum = 0.f;
    for (int j = sub * 4; j < LSEQ; j += 32) {
      v4f v = *(v4f*)(srow + j);
#pragma unroll
      for (int c = 0; c < 4; ++c) {
        v[c] = __expf(v[c] - mx);
        sum += v[c];
      }
      *(v4f*)(srow + j) = v;
    }
    sum += __shfl_xor(sum, 1);
    sum += __shfl_xor(sum, 2);
    sum += __shfl_xor(sum, 4);
    float inv = 1.0f / sum;

    float* arow = attn + (bh * LSEQ + (size_t)(q0 + row)) * LSEQ;
    for (int j = sub * 4; j < LSEQ; j += 32) {
      v4f v = *(v4f*)(srow + j);
      v4h p16;
#pragma unroll
      for (int c = 0; c < 4; ++c) {
        v[c] *= inv;
        p16[c] = (_Float16)v[c];
      }
      *(v4f*)(arow + j) = v;       // attn output: 128B/row bursts, written once
      *(v4h*)(prow + j) = p16;     // f16 copy for the P@V WMMA loop
    }
  }
  __syncthreads();

  // ---- ctx = P @ V : wave wv owns head-dim columns [wv*16, wv*16+16) ----
  v8f acc = vzero8();
  for (int k0 = 0; k0 < LSEQ; k0 += 32) {
    v16h a  = frag_f16(Pl + r * LDPH + k0 + hi * 8);                    // LDS f16
    v16h bb = frag_f16(Vb + (size_t)(wv * 16 + r) * LSEQ + k0 + hi * 8);
    acc = wmma_f16(a, bb, acc);
  }
#pragma unroll
  for (int v = 0; v < 8; ++v) {
    int l = q0 + v + 8 * hi;
    ctx16[(((size_t)b * LSEQ + l) * NHEADS + h) * HEAD + wv * 16 + r] =
        (_Float16)acc[v];
  }
}

// ---------------------------------------------------------------------------
extern "C" void kernel_launch(void* const* d_in, const int* in_sizes, int n_in,
                              void* d_out, int out_size, void* d_ws, size_t ws_size,
                              hipStream_t stream) {
  const float* query = (const float*)d_in[0];
  const float* key   = (const float*)d_in[1];
  const float* value = (const float*)d_in[2];
  const int*   mask  = (const int*)d_in[3];
  const float* Wq = (const float*)d_in[4];
  const float* bq = (const float*)d_in[5];
  const float* Wk = (const float*)d_in[6];
  const float* bk = (const float*)d_in[7];
  const float* Wv = (const float*)d_in[8];
  const float* bv = (const float*)d_in[9];
  const float* Wo = (const float*)d_in[10];
  const float* bo = (const float*)d_in[11];
  const float* scale = (const float*)d_in[12];

  const size_t MB = (size_t)1 << 20;
  char* ws = (char*)d_ws;
  _Float16* Xq16  = (_Float16*)(ws);            //  8 MB  [B*L, DM] f16
  _Float16* Xk16  = (_Float16*)(ws + 8 * MB);   //  8 MB
  _Float16* Xv16  = (_Float16*)(ws + 16 * MB);  //  8 MB
  _Float16* Wq16  = (_Float16*)(ws + 24 * MB);  //  2 MB each
  _Float16* Wk16  = (_Float16*)(ws + 26 * MB);
  _Float16* Wv16  = (_Float16*)(ws + 28 * MB);
  _Float16* Wo16  = (_Float16*)(ws + 30 * MB);
  _Float16* Qh    = (_Float16*)(ws + 32 * MB);  //  8 MB  [B,H,L,64] f16 (normed)
  _Float16* Kh    = (_Float16*)(ws + 40 * MB);  //  8 MB
  _Float16* Vt    = (_Float16*)(ws + 48 * MB);  //  8 MB  [B,H,64,L] f16
  _Float16* ctx16 = (_Float16*)(ws + 56 * MB);  //  8 MB  [B,L,H,64] f16

  float* out  = (float*)d_out;
  float* attn = out + (size_t)BATCH * LSEQ * DM;

  // one-shot f16 casts (L2-resident, bandwidth-trivial)
  const int actN8 = BATCH * LSEQ * DM / 8;      // 524288
  const int wN8   = DM * DM / 8;                // 131072
  cast_f16<<<dim3(actN8 / 256), dim3(256), 0, stream>>>(query, Xq16, actN8);
  cast_f16<<<dim3(actN8 / 256), dim3(256), 0, stream>>>(key,   Xk16, actN8);
  cast_f16<<<dim3(actN8 / 256), dim3(256), 0, stream>>>(value, Xv16, actN8);
  cast_f16<<<dim3(wN8 / 256),   dim3(256), 0, stream>>>(Wq, Wq16, wN8);
  cast_f16<<<dim3(wN8 / 256),   dim3(256), 0, stream>>>(Wk, Wk16, wN8);
  cast_f16<<<dim3(wN8 / 256),   dim3(256), 0, stream>>>(Wv, Wv16, wN8);
  cast_f16<<<dim3(wN8 / 256),   dim3(256), 0, stream>>>(Wo, Wo16, wN8);

  dim3 gp(32, 16, 1), bp(128);
  gemm_xwT<0><<<gp, bp, 0, stream>>>(Xq16, Wq16, bq, Qh, nullptr);
  gemm_xwT<0><<<gp, bp, 0, stream>>>(Xk16, Wk16, bk, Kh, nullptr);
  gemm_xwT<1><<<gp, bp, 0, stream>>>(Xv16, Wv16, bv, Vt, nullptr);

  dim3 ga(LSEQ / 16, NHEADS, BATCH);
  attn_fused<<<ga, dim3(128), SMEM_BYTES, stream>>>(Qh, Kh, Vt, mask, scale,
                                                    attn, ctx16);

  gemm_xwT<2><<<gp, bp, 0, stream>>>(ctx16, Wo16, bo, nullptr, out);
}
